// MoEFFN_53730040873133
// MI455X (gfx1250) — compile-verified, matrix operand order checked by reading, not run on hardware
//
#include <hip/hip_runtime.h>
#include <hip/hip_bf16.h>
#include <math.h>

// ---------------------------------------------------------------------------
// MoE FFN (GPT-2 style, 8 experts, top-2) for gfx1250 (MI455X, wave32, WMMA)
//   out = sum_e cw[n,e] * ( gelu(x @ w1[e]^T + b1[e]) @ w2[e]^T + b2[e] )
// bf16 V_WMMA_F32_16X16X32_BF16, f32 accumulation. Weights pre-converted to
// bf16 in workspace when it fits (halves L2 traffic, kills cvt VALU in the
// hot loop); falls back to on-the-fly f32->bf16 conversion otherwise.
// ---------------------------------------------------------------------------

typedef __attribute__((ext_vector_type(16))) __bf16 v16bf;
typedef __attribute__((ext_vector_type(8)))  __bf16 v8bf;
typedef __attribute__((ext_vector_type(4)))  __bf16 v4bf;
typedef __attribute__((ext_vector_type(8)))  float  v8f;
typedef __attribute__((ext_vector_type(4)))  float  v4f;

#define C_DIM 1024
#define H_DIM 4096
#define NEXP  8
#define TM    64      // token tile per block (B-fragment reuse = TM/16 = 4)
#define HC    64      // H-chunk per iteration

__device__ __forceinline__ v8f wmma_bf16(v16bf a, v16bf b, v8f c) {
  return __builtin_amdgcn_wmma_f32_16x16x32_bf16(
      false, a, false, b, (short)0, c, false, false);
}

// Fragment loaders. ISA 16-bit A/B layout: a lane holds K = {k0..k0+7,
// k0+16..k0+23} with k0 = (lane>=16 ? 8 : 0) + K-step base.
__device__ __forceinline__ v16bf frag_w(const float* __restrict__ p, int k0) {
  v4f a = *(const v4f*)(p + k0);
  v4f b = *(const v4f*)(p + k0 + 4);
  v4f c = *(const v4f*)(p + k0 + 16);
  v4f d = *(const v4f*)(p + k0 + 20);
  v16bf r;
#pragma unroll
  for (int i = 0; i < 4; ++i) {
    r[i]      = (__bf16)a[i];
    r[i + 4]  = (__bf16)b[i];
    r[i + 8]  = (__bf16)c[i];
    r[i + 12] = (__bf16)d[i];
  }
  return r;
}

__device__ __forceinline__ v16bf frag_w(const __bf16* p, int k0) {
  v8bf lo = *(const v8bf*)(p + k0);
  v8bf hi = *(const v8bf*)(p + k0 + 16);
  v16bf r;
#pragma unroll
  for (int i = 0; i < 8; ++i) { r[i] = lo[i]; r[i + 8] = hi[i]; }
  return r;
}

__device__ __forceinline__ float newgelu(float v) {
  return 0.5f * v *
         (1.0f + tanhf(0.7978845608028654f * (v + 0.044715f * v * v * v)));
}

// ---------------------------------------------------------------------------
// Weight pre-conversion: f32 -> bf16, vectorized (float4 in, 4x bf16 out).
// ---------------------------------------------------------------------------
__global__ __launch_bounds__(256) void cvt_f32_bf16(
    const float* __restrict__ s, __bf16* __restrict__ d, long n4) {
  long i = (long)blockIdx.x * 256 + threadIdx.x;
  if (i >= n4) return;
  v4f v = *(const v4f*)(s + i * 4);
  v4bf o;
#pragma unroll
  for (int q = 0; q < 4; ++q) o[q] = (__bf16)v[q];
  *(v4bf*)(d + i * 4) = o;
}

// ---------------------------------------------------------------------------
// Router: one wave32 per token -> cw[N][8] combine weights.
// ---------------------------------------------------------------------------
__global__ __launch_bounds__(256) void moe_router(
    const float* __restrict__ x, const float* __restrict__ rw,
    float* __restrict__ cw, int N) {
  __shared__ float rws[NEXP][C_DIM];  // 32 KB
  const int tid = threadIdx.x;
  for (int i = tid; i < NEXP * C_DIM; i += 256)
    rws[i >> 10][i & (C_DIM - 1)] = rw[i];
  __syncthreads();

  const int wave = tid >> 5;
  const int lane = tid & 31;
  const int n = blockIdx.x * 8 + wave;
  if (n >= N) return;

  const float* xr = x + (size_t)n * C_DIM;
  float p[NEXP];
#pragma unroll
  for (int e = 0; e < NEXP; ++e) p[e] = 0.0f;
  for (int c = lane; c < C_DIM; c += 32) {
    float xv = xr[c];
#pragma unroll
    for (int e = 0; e < NEXP; ++e) p[e] += xv * rws[e][c];
  }
#pragma unroll
  for (int off = 16; off > 0; off >>= 1)
#pragma unroll
    for (int e = 0; e < NEXP; ++e) p[e] += __shfl_xor(p[e], off, 32);

  if (lane == 0) {
    float mx = p[0];
#pragma unroll
    for (int e = 1; e < NEXP; ++e) mx = fmaxf(mx, p[e]);
    float ex[NEXP], s = 0.0f;
#pragma unroll
    for (int e = 0; e < NEXP; ++e) { ex[e] = __expf(p[e] - mx); s += ex[e]; }
    int i0 = 0; float v0 = ex[0];
#pragma unroll
    for (int e = 1; e < NEXP; ++e) if (ex[e] > v0) { v0 = ex[e]; i0 = e; }
    int i1 = -1; float v1 = -1.0f;
#pragma unroll
    for (int e = 0; e < NEXP; ++e)
      if (e != i0 && ex[e] > v1) { v1 = ex[e]; i1 = e; }
    float q0 = v0 / s, q1 = v1 / s;
    float inv = 1.0f / (q0 + q1 + 1e-8f);
    float* cwn = cw + (size_t)n * NEXP;
#pragma unroll
    for (int e = 0; e < NEXP; ++e)
      cwn[e] = (e == i0) ? q0 * inv : (e == i1) ? q1 * inv : 0.0f;
  }
}

// ---------------------------------------------------------------------------
// Fused expert FFN. Block = 256 threads (8 waves), 64-token tile, full
// C=1024 output width. Gate weight cw folded into htile (no atomics).
// GEMM1: wave w owns htile frags (m = w&3, nh = 2*(w>>2)+{0,1})
// GEMM2: wave w owns output columns [w*128, (w+1)*128), acc 4x8 fragments
// ---------------------------------------------------------------------------
template <typename WT>
__global__ __launch_bounds__(256) void moe_ffn(
    const float* __restrict__ x,  const WT* __restrict__ w1,
    const float* __restrict__ b1, const WT* __restrict__ w2,
    const float* __restrict__ b2, const float* __restrict__ cw,
    float* __restrict__ out) {
  __shared__ __align__(16) __bf16 xs[TM][C_DIM];  // 128 KB x-tile (bf16)
  __shared__ __align__(16) __bf16 hs[TM][HC];     //   8 KB gelu'd htile
  __shared__ float cws[TM][NEXP];                 //   2 KB gate weights

  const int tid  = threadIdx.x;
  const int wave = tid >> 5;
  const int lane = tid & 31;
  const int kh   = lane >> 4;
  const int l15  = lane & 15;
  const int t0   = blockIdx.x * TM;

  // ---- Phase 0: stage x tile (f32 -> bf16) and cw tile into LDS ----
  {
    const float* xb = x + (size_t)t0 * C_DIM;
    __bf16* xsf = &xs[0][0];
    for (int i = tid * 4; i < TM * C_DIM; i += 256 * 4) {
      v4f v = *(const v4f*)(xb + i);
#pragma unroll
      for (int q = 0; q < 4; ++q) xsf[i + q] = (__bf16)v[q];
    }
    const float* cwb = cw + (size_t)t0 * NEXP;
    for (int i = tid; i < TM * NEXP; i += 256) (&cws[0][0])[i] = cwb[i];
  }
  __syncthreads();

  v8f acc2[4][8];
#pragma unroll
  for (int m = 0; m < 4; ++m)
#pragma unroll
    for (int nc = 0; nc < 8; ++nc)
#pragma unroll
      for (int i = 0; i < 8; ++i) acc2[m][nc][i] = 0.0f;

  const int m1    = wave & 3;          // GEMM1 token-tile of this wave
  const int nh1   = (wave >> 2) * 2;   // GEMM1 first h-col tile (pair)
  const int cbase = wave * 128;        // GEMM2 output-column base

  for (int e = 0; e < NEXP; ++e) {
    const WT* w1p = w1 + (size_t)e * H_DIM * C_DIM;
    const WT* w2p = w2 + (size_t)e * C_DIM * H_DIM;
    const float* b1p = b1 + (size_t)e * H_DIM;

    for (int hb = 0; hb < H_DIM; hb += HC) {
      // ---- GEMM1: htile[64xHC] = x[64x1024] @ w1_chunk^T ----
      const int h0 = hb + nh1 * 16 + l15;
      const WT* w1row0 = w1p + (size_t)h0 * C_DIM;
      const WT* w1row1 = w1row0 + (size_t)16 * C_DIM;
      __builtin_prefetch(w1row0 + (size_t)32 * C_DIM, 0, 0);
      const __bf16* xrow = &xs[m1 * 16 + l15][0];

      v8f acc1a, acc1b;
#pragma unroll
      for (int i = 0; i < 8; ++i) { acc1a[i] = 0.0f; acc1b[i] = 0.0f; }

#pragma unroll 4
      for (int kk = 0; kk < C_DIM; kk += 32) {
        const int k0 = kk + kh * 8;
        v16bf a  = frag_w(xrow, k0);
        v16bf b0 = frag_w(w1row0, k0);
        v16bf b1f = frag_w(w1row1, k0);
        acc1a = wmma_bf16(a, b0, acc1a);
        acc1b = wmma_bf16(a, b1f, acc1b);
      }

      const float bb0 = b1p[hb + nh1 * 16 + l15];
      const float bb1 = b1p[hb + nh1 * 16 + 16 + l15];
#pragma unroll
      for (int j = 0; j < 8; ++j) {
        const int row = m1 * 16 + j + kh * 8;
        const float g0 = newgelu(acc1a[j] + bb0);
        const float g1 = newgelu(acc1b[j] + bb1);
        const float cwv = cws[row][e];
        hs[row][nh1 * 16 + l15]      = (__bf16)(g0 * cwv);
        hs[row][nh1 * 16 + 16 + l15] = (__bf16)(g1 * cwv);
      }
      __syncthreads();

      // ---- GEMM2: acc2 += htile[64xHC] @ w2_chunk^T ----
#pragma unroll
      for (int kk2 = 0; kk2 < HC; kk2 += 32) {
        const int k0 = kk2 + kh * 8;
        v16bf A[4];
#pragma unroll
        for (int m = 0; m < 4; ++m) A[m] = frag_w(&hs[m * 16 + l15][0], k0);
#pragma unroll
        for (int nc = 0; nc < 8; ++nc) {
          const WT* w2row = w2p + (size_t)(cbase + nc * 16 + l15) * H_DIM;
          v16bf b = frag_w(w2row, hb + k0);
#pragma unroll
          for (int m = 0; m < 4; ++m) acc2[m][nc] = wmma_bf16(A[m], b, acc2[m][nc]);
        }
      }
      __syncthreads();  // protect hs before next chunk rewrites it
    }
  }

  // ---- Epilogue: out = acc2 + sum_e cw[n,e] * b2[e][c] ----
  float* outb = out + (size_t)t0 * C_DIM;
#pragma unroll
  for (int nc = 0; nc < 8; ++nc) {
    const int col = cbase + nc * 16 + l15;
    float b2v[NEXP];
#pragma unroll
    for (int e = 0; e < NEXP; ++e) b2v[e] = b2[(size_t)e * C_DIM + col];
#pragma unroll
    for (int m = 0; m < 4; ++m) {
#pragma unroll
      for (int j = 0; j < 8; ++j) {
        const int row = m * 16 + j + kh * 8;
        float bias = 0.0f;
#pragma unroll
        for (int e = 0; e < NEXP; ++e) bias += cws[row][e] * b2v[e];
        outb[(size_t)row * C_DIM + col] = acc2[m][nc][j] + bias;
      }
    }
  }
}

// ---------------------------------------------------------------------------
extern "C" void kernel_launch(void* const* d_in, const int* in_sizes, int n_in,
                              void* d_out, int out_size, void* d_ws,
                              size_t ws_size, hipStream_t stream) {
  const float* x        = (const float*)d_in[0];
  const float* router_w = (const float*)d_in[1];
  const float* w1       = (const float*)d_in[2];
  const float* b1       = (const float*)d_in[3];
  const float* w2       = (const float*)d_in[4];
  const float* b2       = (const float*)d_in[5];
  float* out = (float*)d_out;

  const int N = in_sizes[0] / C_DIM;  // B*T tokens

  // Workspace layout: [cw: N*8 f32][w1 bf16][w2 bf16]
  float* cw = (float*)d_ws;
  const size_t cwBytes = ((size_t)N * NEXP * sizeof(float) + 255) & ~(size_t)255;
  const size_t wElems  = (size_t)NEXP * H_DIM * C_DIM;
  const size_t need    = cwBytes + 2 * wElems * sizeof(__bf16);

  moe_router<<<dim3((N + 7) / 8), dim3(256), 0, stream>>>(x, router_w, cw, N);

  if (ws_size >= need) {
    // Pre-convert weights to bf16 once; hot loop then streams half the bytes
    // (entire bf16 weight set fits in the 192MB L2) with no cvt VALU.
    __bf16* w1b = (__bf16*)((char*)d_ws + cwBytes);
    __bf16* w2b = w1b + wElems;
    const long n4 = (long)(wElems / 4);
    const int cvtBlocks = (int)((n4 + 255) / 256);
    cvt_f32_bf16<<<dim3(cvtBlocks), dim3(256), 0, stream>>>(w1, w1b, n4);
    cvt_f32_bf16<<<dim3(cvtBlocks), dim3(256), 0, stream>>>(w2, w2b, n4);
    moe_ffn<__bf16><<<dim3(N / TM), dim3(256), 0, stream>>>(
        x, w1b, b1, w2b, b2, cw, out);
  } else {
    moe_ffn<float><<<dim3(N / TM), dim3(256), 0, stream>>>(
        x, w1, b1, w2, b2, cw, out);
  }
}
